// SinkhornNet_15324443312525
// MI455X (gfx1250) — matrix operand order repaired and checked by reading, not compiled, not run on hardware
//
#include <hip/hip_runtime.h>

// ---------------------------------------------------------------------------
// SinkhornNet on MI455X (gfx1250, wave32)
//   dsm = Sinkhorn(weight, 20 steps);  out = dsm @ x   (512 x 65536, fp32)
//
// Sinkhorn reduced to vector iterations: E = exp(W); u=1;
//   repeat 20x: v = 1/(E u)   (row normalize)
//              u = 1/(E^T v)  (col normalize)
//   dsm_ij = v_i * E_ij * u_j
// GEMM: split-bf16 (hi+lo) x3 WMMA for ~fp32 accuracy at bf16 matrix rate.
// A-plane staging into LDS via the Tensor Data Mover (tensor_load_to_lds)
// with hardware LDS padding; fragments read as ds_load_b128.
// ---------------------------------------------------------------------------

#define NDIM      512
#define NSTEPS    20
#define NWG_SINK  32
#define ROWS_PER_WG (NDIM / NWG_SINK)   // 16 rows per workgroup

typedef __bf16 bf16_t;
typedef bf16_t   v16bf __attribute__((ext_vector_type(16)));
typedef float    v8f   __attribute__((ext_vector_type(8)));
typedef unsigned u32x4 __attribute__((ext_vector_type(4)));
typedef unsigned v4u   __attribute__((ext_vector_type(4)));
typedef int      v8i   __attribute__((ext_vector_type(8)));
typedef int      v4i   __attribute__((ext_vector_type(4)));

#if __has_builtin(__builtin_amdgcn_tensor_load_to_lds) && \
    __has_builtin(__builtin_amdgcn_s_wait_tensorcnt)
#define USE_TDM 1
#else
#define USE_TDM 0
#endif

static __device__ __forceinline__ unsigned short f32_to_bf16(float f) {
  unsigned u = __float_as_uint(f);
  unsigned r = 0x7FFFu + ((u >> 16) & 1u);     // round-to-nearest-even
  return (unsigned short)((u + r) >> 16);
}
static __device__ __forceinline__ float bf16_to_f32(unsigned short h) {
  return __uint_as_float((unsigned)h << 16);
}
// byte offset of a __shared__ object within the LDS segment
static __device__ __forceinline__ unsigned lds_offset(const void* p) {
  return (unsigned)(unsigned long long)
      (__attribute__((address_space(3))) const char*)p;
}

// ---------------- device-wide sense barrier (32 resident WGs) --------------
__device__ __forceinline__ void gbarrier(unsigned* bar) {
  // bar[0] = arrival counter (self-resetting), bar[1] = generation
  __syncthreads();
  if (threadIdx.x == 0) {
    unsigned gen = __hip_atomic_load(&bar[1], __ATOMIC_RELAXED,
                                     __HIP_MEMORY_SCOPE_AGENT);
    unsigned arrived = __hip_atomic_fetch_add(&bar[0], 1u, __ATOMIC_ACQ_REL,
                                              __HIP_MEMORY_SCOPE_AGENT) + 1u;
    if (arrived == NWG_SINK) {
      __hip_atomic_store(&bar[0], 0u, __ATOMIC_RELAXED,
                         __HIP_MEMORY_SCOPE_AGENT);
      __hip_atomic_fetch_add(&bar[1], 1u, __ATOMIC_ACQ_REL,
                             __HIP_MEMORY_SCOPE_AGENT);
    } else {
      while (__hip_atomic_load(&bar[1], __ATOMIC_ACQUIRE,
                               __HIP_MEMORY_SCOPE_AGENT) == gen)
        __builtin_amdgcn_s_sleep(1);
    }
  }
  __syncthreads();
}

// v_out[row] = 1 / dot(M[row,:], v_in)   -- one wave handles 2 rows
static __device__ __forceinline__ void sink_pass(const float* __restrict__ M,
                                                 const float* __restrict__ vin,
                                                 float* __restrict__ vout,
                                                 int wg, int tid) {
  const int wave = tid >> 5, lane = tid & 31;
  for (int rr = 0; rr < 2; ++rr) {
    const int row = wg * ROWS_PER_WG + wave * 2 + rr;
    const float* mr = M + row * NDIM;
    float s = 0.0f;
    for (int k = lane; k < NDIM; k += 32) s = fmaf(mr[k], vin[k], s);
    for (int off = 16; off > 0; off >>= 1) s += __shfl_xor(s, off, 32);
    if (lane == 0) vout[row] = 1.0f / s;
  }
}

__global__ void __launch_bounds__(256)
sinkhorn_kernel(const float* __restrict__ W, float* __restrict__ E,
                float* __restrict__ ET, float* __restrict__ u,
                float* __restrict__ v, unsigned short* __restrict__ Ahi,
                unsigned short* __restrict__ Alo, unsigned* bar) {
  const int wg = blockIdx.x, tid = threadIdx.x;

  // Phase A: E = exp(W), ET = E^T, u = 1
  for (int idx = tid; idx < ROWS_PER_WG * NDIM; idx += 256) {
    const int row = wg * ROWS_PER_WG + (idx >> 9);
    const int col = idx & (NDIM - 1);
    const float e = __expf(W[row * NDIM + col]);
    E[row * NDIM + col] = e;
    ET[col * NDIM + row] = e;
  }
  if (wg == 0)
    for (int j = tid; j < NDIM; j += 256) u[j] = 1.0f;
  gbarrier(bar);

  // Phase B: 20 x (row normalize, col normalize) as matvec + reciprocal
  for (int it = 0; it < NSTEPS; ++it) {
    sink_pass(E,  u, v, wg, tid);  gbarrier(bar);   // v = 1/(E u)
    sink_pass(ET, v, u, wg, tid);  gbarrier(bar);   // u = 1/(E^T v)
  }

  // Phase C: dsm = diag(v) E diag(u), materialized as bf16 hi + lo planes
  for (int idx = tid; idx < ROWS_PER_WG * NDIM; idx += 256) {
    const int row = wg * ROWS_PER_WG + (idx >> 9);
    const int col = idx & (NDIM - 1);
    const float a = E[row * NDIM + col] * v[row] * u[col];
    const unsigned short hi = f32_to_bf16(a);
    Ahi[row * NDIM + col] = hi;
    Alo[row * NDIM + col] = f32_to_bf16(a - bf16_to_f32(hi));
  }
}

// ---------------------------------------------------------------------------
// GEMM: out[512][N] = dsm @ x, split-bf16 3-WMMA.
// Block 256 thr = 8 waves (4M x 2N), per-wave 32x32 out, BM=128 BN=64 KC=32.
// LDS rows padded to 20 dwords so fragments are aligned ds_load_b128.
// ---------------------------------------------------------------------------
#define BM 128
#define BN 64
#define KC 32
#define LDKD 20         // padded LDS row stride in dwords (= 40 bf16, 80 B)

__global__ void __launch_bounds__(256)
wmma_gemm_kernel(const float* __restrict__ x,
                 const unsigned short* __restrict__ Ahi,
                 const unsigned short* __restrict__ Alo,
                 float* __restrict__ out, int N) {
  __shared__ __align__(16) unsigned As[2][BM * LDKD];  // [plane][row*20 + dw]
  __shared__ __align__(16) unsigned Bs[2][BN * LDKD];  // [plane][col*20 + dw]

  const int tid  = threadIdx.x;
  const int wave = tid >> 5, lane = tid & 31;
  const int mw = wave >> 1, nw = wave & 1;      // 4 x 2 wave grid
  const int lm = lane & 15, half = lane >> 4;
  const int Mb = blockIdx.y * BM;
  const int Nb = blockIdx.x * BN;

  v8f acc[2][2] = {};

#if !USE_TDM
  const unsigned* gAhi = (const unsigned*)Ahi;  // dsm planes, row-major bf16
  const unsigned* gAlo = (const unsigned*)Alo;
#endif

  for (int kt = 0; kt < NDIM; kt += KC) {
#if USE_TDM
    // Tensor Data Mover: 32x128 (2-byte) tile from 512-stride tensor into
    // LDS with hw padding: +4 dwords after every 16 dwords -> 20 dword rows.
    if (wave == 0) {
      for (int p = 0; p < 2; ++p) {
        const unsigned short* src = (p == 0) ? Ahi : Alo;
        const unsigned long long ga =
            (unsigned long long)(const void*)(src + (size_t)Mb * NDIM + kt);
        v4u g0;
        g0[0] = 1u;                                   // count=1, user D#
        g0[1] = lds_offset(&As[p][0]);                // lds_addr (bytes)
        g0[2] = (unsigned)ga;                         // global_addr[31:0]
        g0[3] = (unsigned)(ga >> 32) | (2u << 30);    // addr[56:32] | type=2
        v8i g1;
        g1[0] = (int)((1u << 16)      // data_size = 2 bytes
                      | (1u << 20)    // pad_enable
                      | (3u << 22)    // pad_interval: 16 dwords
                      | (3u << 25));  // pad_amount: 4 dwords
        g1[1] = (int)(512u << 16);    // tensor_dim0 = 512 (low 16 -> [31:16])
        g1[2] = (int)(512u << 16);    // dim0 hi=0 | tensor_dim1=512 low16
        g1[3] = (int)(32u << 16);     // dim1 hi=0 | tile_dim0 = 32
        g1[4] = 128;                  // tile_dim1 = 128 | tile_dim2 = 0
        g1[5] = 512;                  // tensor_dim0_stride = 512 (low 32)
        g1[6] = 0;                    // stride hi | tensor_dim1_stride lo
        g1[7] = 0;
        v4i g2 = {0, 0, 0, 0};        // unused (2D tile)
        v4i g3 = {0, 0, 0, 0};
        v8i g4 = {0, 0, 0, 0, 0, 0, 0, 0};  // unused extension group
        __builtin_amdgcn_tensor_load_to_lds(g0, g1, g2, g3, g4, 0);
      }
      __builtin_amdgcn_s_wait_tensorcnt((short)0);
    }
#else
    // Fallback: stage A chunk (128x32 bf16, both planes) as dwords
    for (int i = 0; i < 8; ++i) {
      const int linear = tid + i * 256;          // 0..2047
      const int row = linear >> 4, dw = linear & 15;
      const int g = (Mb + row) * (NDIM / 2) + (kt >> 1) + dw;
      As[0][row * LDKD + dw] = gAhi[g];
      As[1][row * LDKD + dw] = gAlo[g];
    }
#endif
    // Stage X chunk (32x64 fp32), split to bf16 hi/lo, store K-contiguous
    {
      const int col = tid & 63, k0 = tid >> 6;
      unsigned short* bh = (unsigned short*)Bs[0];
      unsigned short* bl = (unsigned short*)Bs[1];
      for (int i = 0; i < 8; ++i) {
        const int k = k0 + i * 4;
        const float xv = x[(size_t)(kt + k) * N + (Nb + col)];
        const unsigned short hi = f32_to_bf16(xv);
        bh[col * (2 * LDKD) + k] = hi;
        bl[col * (2 * LDKD) + k] = f32_to_bf16(xv - bf16_to_f32(hi));
      }
    }
    __syncthreads();

    union Frag { v16bf v; u32x4 q[2]; };
    Frag a[2][2], b[2][2];
    // A fragment (ISA layout): lane m, half h; vgpr j holds
    //   K = (j>>2)*16 + h*8 + (j&3)*2  ->  dwords [h*4 .. h*4+3] and [8+h*4 ..]
    for (int mi = 0; mi < 2; ++mi) {
      const int row = mw * 32 + mi * 16 + lm;
      for (int p = 0; p < 2; ++p) {
        a[mi][p].q[0] = *(const u32x4*)&As[p][row * LDKD + half * 4];
        a[mi][p].q[1] = *(const u32x4*)&As[p][row * LDKD + 8 + half * 4];
      }
    }
    // B fragment: lane n, half h; same K layout as A
    //   -> dwords [h*4 .. h*4+3] and [8+h*4 ..]
    for (int ni = 0; ni < 2; ++ni) {
      const int col = nw * 32 + ni * 16 + lm;
      for (int p = 0; p < 2; ++p) {
        b[ni][p].q[0] = *(const u32x4*)&Bs[p][col * LDKD + half * 4];
        b[ni][p].q[1] = *(const u32x4*)&Bs[p][col * LDKD + 8 + half * 4];
      }
    }
    // out += Ahi*Bhi + Ahi*Blo + Alo*Bhi  (drop lo*lo)
    for (int mi = 0; mi < 2; ++mi)
      for (int ni = 0; ni < 2; ++ni) {
        acc[mi][ni] = __builtin_amdgcn_wmma_f32_16x16x32_bf16(
            false, a[mi][0].v, false, b[ni][0].v, (short)0, acc[mi][ni],
            false, false);
        acc[mi][ni] = __builtin_amdgcn_wmma_f32_16x16x32_bf16(
            false, a[mi][0].v, false, b[ni][1].v, (short)0, acc[mi][ni],
            false, false);
        acc[mi][ni] = __builtin_amdgcn_wmma_f32_16x16x32_bf16(
            false, a[mi][1].v, false, b[ni][0].v, (short)0, acc[mi][ni],
            false, false);
      }
    __syncthreads();
  }

  // C/D layout: vgpr r -> M = r + 8*half, N = lane&15
  for (int mi = 0; mi < 2; ++mi)
    for (int ni = 0; ni < 2; ++ni) {
      const int row0 = Mb + mw * 32 + mi * 16 + half * 8;
      const int col  = Nb + nw * 32 + ni * 16 + lm;
      for (int r = 0; r < 8; ++r)
        out[(size_t)(row0 + r) * N + col] = acc[mi][ni][r];
    }
}

// ---------------------------------------------------------------------------
extern "C" void kernel_launch(void* const* d_in, const int* in_sizes, int n_in,
                              void* d_out, int out_size, void* d_ws,
                              size_t ws_size, hipStream_t stream) {
  const float* x = (const float*)d_in[0];       // [512][65536] fp32
  const float* w = (const float*)d_in[1];       // [512][512]   fp32
  float* out = (float*)d_out;                   // [512][65536] fp32
  const int N = in_sizes[0] / NDIM;             // 65536

  // Workspace layout (~3.1 MB needed)
  char* ws = (char*)d_ws;
  unsigned* bar = (unsigned*)ws;                               // 64 B
  float* u  = (float*)(ws + 1024);                             // 2 KB
  float* v  = (float*)(ws + 3072);                             // 2 KB
  float* E  = (float*)(ws + 8192);                             // 1 MB
  float* ET = E + NDIM * NDIM;                                 // 1 MB
  unsigned short* Ahi =
      (unsigned short*)(ws + 8192 + 2u * NDIM * NDIM * 4u);    // 512 KB
  unsigned short* Alo = Ahi + NDIM * NDIM;                     // 512 KB

  (void)hipMemsetAsync(bar, 0, 64, stream);     // reset barrier each replay

  sinkhorn_kernel<<<NWG_SINK, 256, 0, stream>>>(w, E, ET, u, v, Ahi, Alo, bar);

  dim3 grid(N / BN, NDIM / BM);                 // (1024, 4)
  wmma_gemm_kernel<<<grid, 256, 0, stream>>>(x, Ahi, Alo, out, N);
}
